// SSIMLoss_18408229830674
// MI455X (gfx1250) — matrix operand order repaired and checked by reading, not compiled, not run on hardware
//
#include <hip/hip_runtime.h>

// ---------------------------------------------------------------------------
// Fused SSIM loss for MI455X (gfx1250, wave32, WMMA).
//
// Roofline: min traffic = 100.6 MB (pred+target) -> 4.3 us @ 23.3 TB/s.
// Scalar-f32 blur needs ~28 FLOP/byte -> f32 VALU would be compute-bound, and
// the unfused reference moves >1 GB of intermediates. So: single fused kernel,
// 11-tap separable Gaussian expressed as banded-matrix v_wmma_f32_16x16x32_f16
// (26-tap window padded to K=32), f16 inputs / f32 accumulation, LDS tiling
// (halo redundancy 1.52x -> ~152 MB ~ 6.5 us, near the HBM bound).
//
// Per 16x16 output tile per map: 2 WMMAs (horizontal pass, 26 live rows in two
// 16-row blocks) + 1 WMMA (vertical pass) = 3; 5 maps -> 15 WMMAs/tile.
// Banded matrices G[k][n] = g[k-n] (zero outside [0,10]) make all K>=26
// rows/cols zero-weight, so halo garbage beyond the 26-window is harmless.
// ---------------------------------------------------------------------------

typedef _Float16 v16h __attribute__((ext_vector_type(16)));
typedef _Float16 v8h  __attribute__((ext_vector_type(8)));
typedef float    v8f  __attribute__((ext_vector_type(8)));

#define IMG     512
#define REGW    64            // output region width per block
#define REGH    32            // output region height per block
#define SMROWS  48            // REGH + 10 halo, padded to cover K<=31 row reads
#define SMCOLS  88            // REGW + 10 halo, padded: keeps rows 16B-aligned
#define HBCOLS  40            // transposed H buffer row stride (bank-friendly, 16B-aligned)
#define NPLANE  48            // 16 * 3
#define NBLOCKS (NPLANE * (IMG / REGW) * (IMG / REGH))  // 48*8*16 = 6144
#define SSIM_C1 0.0001f
#define SSIM_C2 0.0009f
#define NPIX    12582912.0    // 16*3*512*512

// 11-tap Gaussian, sigma=1.5, normalized (matches reference _gauss1d()).
__device__ __constant__ float GW[11] = {
    0.00102838f, 0.00759876f, 0.03600077f, 0.10936068f, 0.21300553f,
    0.26601172f,
    0.21300553f, 0.10936068f, 0.03600077f, 0.00759876f, 0.00102838f};

__device__ __forceinline__ float gw_or_zero(int d) {
  int dc = d < 0 ? 0 : (d > 10 ? 10 : d);
  float w = GW[dc];
  return (d >= 0 && d <= 10) ? w : 0.0f;
}

__global__ __launch_bounds__(256) void ssim_fused(const float* __restrict__ pred,
                                                  const float* __restrict__ targ,
                                                  float* __restrict__ partial) {
  __shared__ __align__(16) _Float16 sm[5][SMROWS][SMCOLS];   // p, t, p*p, t*t, p*t
  __shared__ __align__(16) _Float16 hbT[8][16][HBCOLS];      // per-wave mid buffer, transposed
  __shared__ float red[256];

  const int tid  = threadIdx.x;
  const int lane = tid & 31;
  const int wv   = tid >> 5;
  const int b    = blockIdx.x;
  const int plane = b >> 7;          // 128 blocks per plane
  const int rem   = b & 127;
  const int by    = rem >> 3;        // 0..15 (rows of 32)
  const int bx    = rem & 7;         // 0..7  (cols of 64)
  const size_t base = (size_t)plane * IMG * IMG;

  // ---- Stage 0: halo tile of the 5 maps -> LDS (f16), zero-padded outside image.
  for (int idx = tid; idx < SMROWS * SMCOLS; idx += 256) {
    int r = idx / SMCOLS;
    int c = idx - r * SMCOLS;
    int gy = by * REGH - 5 + r;
    int gx = bx * REGW - 5 + c;
    float p = 0.0f, t = 0.0f;
    if ((unsigned)gy < IMG && (unsigned)gx < IMG) {
      size_t o = base + (size_t)gy * IMG + gx;
      p = pred[o];
      t = targ[o];
    }
    sm[0][r][c] = (_Float16)p;
    sm[1][r][c] = (_Float16)t;
    sm[2][r][c] = (_Float16)(p * p);
    sm[3][r][c] = (_Float16)(t * t);
    sm[4][r][c] = (_Float16)(p * t);
  }
  __syncthreads();

  // ---- Constant banded-Gaussian fragments.
  const bool hiHalf = lane >= 16;
  const int  ln     = lane & 15;
  v16h gcB;  // B-layout 32x16: K = (h&7) + (h>=8?16:0) + (hi?8:0), N = ln, val = g[K-N]
  v16h grA;  // A-layout 16x32: M = ln, K = h + (hi?16:0), val = g[K-M]
#pragma unroll
  for (int h = 0; h < 16; ++h) {
    int kb = (h & 7) + (h >= 8 ? 16 : 0) + (hiHalf ? 8 : 0);
    int ka = h + (hiHalf ? 16 : 0);
    gcB[h] = (_Float16)gw_or_zero(kb - ln);
    grA[h] = (_Float16)gw_or_zero(ka - ln);
  }

  // ---- Per-wave 16x16 tile: tiles are 4 wide x 2 tall in the 64x32 region.
  const int ty = wv >> 2, tx = wv & 3;
  const int row0 = ty * 16, col0 = tx * 16;
  const int acolBase = col0 + (hiHalf ? 8 : 0);

  v8f acc[5];
#pragma unroll
  for (int m = 0; m < 5; ++m) {
    // Pass 1 (horizontal): H[j][n] = sum_k sm[row0+j][col0+k] * g[k-n], j=0..31.
#pragma unroll
    for (int rb = 0; rb < 2; ++rb) {
      const _Float16* rp = &sm[m][row0 + rb * 16 + ln][acolBase];
      v8h a0 = *(const v8h*)rp;          // K run {0..7}/{8..15}
      v8h a1 = *(const v8h*)(rp + 16);   // K run {16..23}/{24..31}
      v16h afrag = __builtin_shufflevector(a0, a1, 0, 1, 2, 3, 4, 5, 6, 7,
                                           8, 9, 10, 11, 12, 13, 14, 15);
      v8f z = {0.f, 0.f, 0.f, 0.f, 0.f, 0.f, 0.f, 0.f};
      v8f hres = __builtin_amdgcn_wmma_f32_16x16x32_f16(
          false, afrag, false, gcB, (short)0, z, false, false);
      // D layout: N = ln, M = i + (hi?8:0); store transposed: hbT[wv][N][row].
#pragma unroll
      for (int i = 0; i < 8; ++i)
        hbT[wv][ln][rb * 16 + i + (hiHalf ? 8 : 0)] = (_Float16)hres[i];
    }
    asm volatile("" ::: "memory");  // keep LDS stores before the reload below

    // Pass 2 (vertical): D[r][n] = sum_j g[j-r] * H[j][n].
    const _Float16* hp = &hbT[wv][ln][hiHalf ? 16 : 0];
    v8h b0 = *(const v8h*)hp;
    v8h b1 = *(const v8h*)(hp + 8);
    v16h bfrag = __builtin_shufflevector(b0, b1, 0, 1, 2, 3, 4, 5, 6, 7,
                                         8, 9, 10, 11, 12, 13, 14, 15);
    v8f z2 = {0.f, 0.f, 0.f, 0.f, 0.f, 0.f, 0.f, 0.f};
    acc[m] = __builtin_amdgcn_wmma_f32_16x16x32_f16(
        false, grA, false, bfrag, (short)0, z2, false, false);
    asm volatile("" ::: "memory");  // next map's stores must stay after this load
  }

  // ---- Elementwise SSIM on the 8 f32 accumulator elements per lane.
  float lsum = 0.0f;
#pragma unroll
  for (int i = 0; i < 8; ++i) {
    float m1 = acc[0][i], m2 = acc[1][i];
    float m1m2 = m1 * m2, m1s = m1 * m1, m2s = m2 * m2;
    float s1  = acc[2][i] - m1s;
    float s2  = acc[3][i] - m2s;
    float s12 = acc[4][i] - m1m2;
    float num = (2.0f * m1m2 + SSIM_C1) * (2.0f * s12 + SSIM_C2);
    float den = (m1s + m2s + SSIM_C1) * (s1 + s2 + SSIM_C2);
    lsum += num / den;
  }

  // ---- Deterministic block tree reduction.
  red[tid] = lsum;
  __syncthreads();
#pragma unroll
  for (int s = 128; s > 0; s >>= 1) {
    if (tid < s) red[tid] += red[tid + s];
    __syncthreads();
  }
  if (tid == 0) partial[b] = red[0];
}

__global__ __launch_bounds__(256) void ssim_finalize(const float* __restrict__ partial,
                                                     float* __restrict__ out) {
  __shared__ float red[256];
  float s = 0.0f;
  for (int i = threadIdx.x; i < NBLOCKS; i += 256) s += partial[i];
  red[threadIdx.x] = s;
  __syncthreads();
#pragma unroll
  for (int k = 128; k > 0; k >>= 1) {
    if (threadIdx.x < k) red[threadIdx.x] += red[threadIdx.x + k];
    __syncthreads();
  }
  if (threadIdx.x == 0) out[0] = (float)(1.0 - (double)red[0] / NPIX);
}

extern "C" void kernel_launch(void* const* d_in, const int* in_sizes, int n_in,
                              void* d_out, int out_size, void* d_ws, size_t ws_size,
                              hipStream_t stream) {
  const float* pred = (const float*)d_in[0];
  const float* targ = (const float*)d_in[1];
  float* out     = (float*)d_out;
  float* partial = (float*)d_ws;  // NBLOCKS floats = 24 KB scratch

  ssim_fused<<<dim3(NBLOCKS), dim3(256), 0, stream>>>(pred, targ, partial);
  ssim_finalize<<<dim3(1), dim3(256), 0, stream>>>(partial, out);
}
